// WaveletMultiHeadAttention_26963804684328
// MI455X (gfx1250) — matrix operand order repaired and checked by reading, not compiled, not run on hardware
//
#include <hip/hip_runtime.h>
#include <hip/hip_bf16.h>

// ---------------------------------------------------------------------------
// Types
// ---------------------------------------------------------------------------
typedef __attribute__((ext_vector_type(16))) __bf16 v16bf;
typedef __attribute__((ext_vector_type(8)))  __bf16 v8bf;
typedef __attribute__((ext_vector_type(4)))  __bf16 v4bf;
typedef __attribute__((ext_vector_type(8)))  float  v8f;

#define INV_SQRT2_F 0.70710678118654752440f

// WGP-scope prefetch (scope 0 = pull into all cache levels on miss).
// __builtin_prefetch only reaches SE/SYS scopes, which skip the WGP cache.
__device__ inline void prefetch_wgp(const void* p) {
  asm volatile("global_prefetch_b8 %0, off" :: "v"(p));
}

// ---------------------------------------------------------------------------
// Kernel 1: f32 -> bf16 conversion (vectorized: 4 elements / thread)
// ---------------------------------------------------------------------------
__global__ __launch_bounds__(256)
void cvt_bf16_kernel(const float* __restrict__ src, __bf16* __restrict__ dst,
                     size_t n4) {
  size_t i = (size_t)blockIdx.x * blockDim.x + threadIdx.x;
  if (i < n4) {
    float4 v = ((const float4*)src)[i];
    v4bf o;
    o[0] = (__bf16)v.x; o[1] = (__bf16)v.y; o[2] = (__bf16)v.z; o[3] = (__bf16)v.w;
    ((v4bf*)dst)[i] = o;
  }
}

// ---------------------------------------------------------------------------
// Kernel 2: weight prep.  wqk = bf16(Wq+Wk), wv = bf16(Wv), bqk = bq+bk
// ---------------------------------------------------------------------------
__global__ __launch_bounds__(256)
void prep_w_kernel(const float* __restrict__ Wq, const float* __restrict__ Wk,
                   const float* __restrict__ Wv, const float* __restrict__ bq,
                   const float* __restrict__ bk,
                   __bf16* __restrict__ wqk, __bf16* __restrict__ wv,
                   float* __restrict__ bqk, size_t n4, int d4) {
  size_t i = (size_t)blockIdx.x * blockDim.x + threadIdx.x;
  if (i < n4) {
    float4 a = ((const float4*)Wq)[i];
    float4 b = ((const float4*)Wk)[i];
    float4 c = ((const float4*)Wv)[i];
    v4bf s, t;
    s[0] = (__bf16)(a.x + b.x); s[1] = (__bf16)(a.y + b.y);
    s[2] = (__bf16)(a.z + b.z); s[3] = (__bf16)(a.w + b.w);
    t[0] = (__bf16)c.x; t[1] = (__bf16)c.y; t[2] = (__bf16)c.z; t[3] = (__bf16)c.w;
    ((v4bf*)wqk)[i] = s;
    ((v4bf*)wv)[i]  = t;
  }
  if (i < (size_t)d4) {
    float4 a = ((const float4*)bq)[i];
    float4 b = ((const float4*)bk)[i];
    ((float4*)bqk)[i] = make_float4(a.x + b.x, a.y + b.y, a.z + b.z, a.w + b.w);
  }
}

// ---------------------------------------------------------------------------
// WMMA fragment loaders (CDNA5 ISA 7.12.2 bf16 layouts, wave32)
//
// A (16x32, row-major source, lane L: row = L&15, h = L>>4):
//   V0..V3 hold K = 8h..8h+7 ; V4..V7 hold K = 16+8h..16+8h+7
// B (32x16, B = W^T so column n of B == row n of W, contiguous):
//   lane L: col = L&15, h = L>>4 ; V0..V3 hold K = 8h..8h+7 ; V4..V7 hold K = 16+8h..16+8h+7
// ---------------------------------------------------------------------------
union FragU { v16bf v; v8bf h2[2]; };

__device__ inline v16bf load_frag2(const __bf16* p, int off0, int off1) {
  FragU u;
  u.h2[0] = *(const v8bf*)(p + off0);
  u.h2[1] = *(const v8bf*)(p + off1);
  return u.v;
}

__device__ inline v8f wmma_bf16(v16bf a, v16bf b, v8f c) {
  return __builtin_amdgcn_wmma_f32_16x16x32_bf16(
      /*neg_a=*/false, a, /*neg_b=*/false, b,
      /*c_mod=*/(short)0, c, /*reuse_a=*/false, /*reuse_b=*/false);
}

// Store one 16x16 f32 C tile + bias.  Lane L: n = L&15, rows = i + 8*(L>>4).
__device__ inline void store_tile(float* __restrict__ C, int ldc,
                                  int m0, int n0, int lane,
                                  const v8f& c, float bval) {
  const int h = lane >> 4, n = lane & 15;
  float* p = C + (size_t)(m0 + 8 * h) * ldc + (n0 + n);
#pragma unroll
  for (int i = 0; i < 8; ++i) p[(size_t)i * ldc] = c[i] + bval;
}

// ---------------------------------------------------------------------------
// Kernel 3: GEMM  C[M,N] = A[M,K](bf16) * W[N,K]^T(bf16) + bias[N]   (f32 out)
// Block: 256 threads = 8 waves arranged 2(M) x 4(N); wave tile 64x32
// (4x2 WMMA tiles -> 8 v_wmma per 32-wide K-step), block tile 128x128.
// K loop steps 64 (two substeps); one WGP-scope prefetch per stream per
// 128-byte line, two lines ahead.
// ---------------------------------------------------------------------------
__global__ __launch_bounds__(256, 2)
void gemm_bf16_wmma_kernel(const __bf16* __restrict__ A,
                           const __bf16* __restrict__ W,
                           const float*  __restrict__ bias,
                           float* __restrict__ C,
                           int M, int N, int K) {
  const int lane = threadIdx.x & 31;
  const int wave = threadIdx.x >> 5;
  const int wn = wave & 3;   // 4 waves along N
  const int wm = wave >> 2;  // 2 waves along M
  const int m0 = blockIdx.y * 128 + wm * 64;
  const int n0 = blockIdx.x * 128 + wn * 32;
  const int h = lane >> 4;
  const int r = lane & 15;

  const __bf16* ap[4];
  const __bf16* bp[2];
#pragma unroll
  for (int mt = 0; mt < 4; ++mt)
    ap[mt] = A + (size_t)(m0 + mt * 16 + r) * K;
#pragma unroll
  for (int nt = 0; nt < 2; ++nt)
    bp[nt] = W + (size_t)(n0 + nt * 16 + r) * K;

  const int aoff0 = 8 * h,  aoff1 = 16 + 8 * h;
  const int boff0 = 8 * h,  boff1 = 16 + 8 * h;

  v8f acc[4][2];
#pragma unroll
  for (int mt = 0; mt < 4; ++mt)
#pragma unroll
    for (int nt = 0; nt < 2; ++nt) acc[mt][nt] = (v8f){};

  for (int k = 0; k < K; k += 64) {
    // prefetch 128B line 256 B ahead in each of the 6 streams into the WGP cache
#pragma unroll
    for (int mt = 0; mt < 4; ++mt) prefetch_wgp(ap[mt] + k + 128);
#pragma unroll
    for (int nt = 0; nt < 2; ++nt) prefetch_wgp(bp[nt] + k + 128);

#pragma unroll
    for (int kk = 0; kk < 64; kk += 32) {
      v16bf af[4], bf[2];
#pragma unroll
      for (int mt = 0; mt < 4; ++mt)
        af[mt] = load_frag2(ap[mt] + k + kk, aoff0, aoff1);
#pragma unroll
      for (int nt = 0; nt < 2; ++nt)
        bf[nt] = load_frag2(bp[nt] + k + kk, boff0, boff1);

#pragma unroll
      for (int mt = 0; mt < 4; ++mt)
#pragma unroll
        for (int nt = 0; nt < 2; ++nt)
          acc[mt][nt] = wmma_bf16(af[mt], bf[nt], acc[mt][nt]);
    }
  }

  float bv[2];
#pragma unroll
  for (int nt = 0; nt < 2; ++nt) bv[nt] = bias[n0 + nt * 16 + r];
#pragma unroll
  for (int mt = 0; mt < 4; ++mt)
#pragma unroll
    for (int nt = 0; nt < 2; ++nt)
      store_tile(C, N, m0 + mt * 16, n0 + nt * 16, lane, acc[mt][nt], bv[nt]);
}

// ---------------------------------------------------------------------------
// Kernel 4: fused DWT -> softmax -> (+vL/vH) -> softmax -> iDWT per row.
// One 256-thread block per row; each thread owns 2 of the 512 half-elements.
// ---------------------------------------------------------------------------
__device__ inline void block_reduce2(float& a, float& b, bool is_max,
                                     float* sm) {
#pragma unroll
  for (int off = 16; off; off >>= 1) {
    float a2 = __shfl_xor(a, off, 32);
    float b2 = __shfl_xor(b, off, 32);
    if (is_max) { a = fmaxf(a, a2); b = fmaxf(b, b2); }
    else        { a += a2;          b += b2;          }
  }
  const int w = threadIdx.x >> 5;
  if ((threadIdx.x & 31) == 0) { sm[2 * w] = a; sm[2 * w + 1] = b; }
  __syncthreads();
  if (threadIdx.x == 0) {
    float ra = sm[0], rb = sm[1];
#pragma unroll
    for (int i = 1; i < 8; ++i) {
      if (is_max) { ra = fmaxf(ra, sm[2 * i]); rb = fmaxf(rb, sm[2 * i + 1]); }
      else        { ra += sm[2 * i];           rb += sm[2 * i + 1];           }
    }
    sm[16] = ra; sm[17] = rb;
  }
  __syncthreads();
  a = sm[16]; b = sm[17];
}

__global__ __launch_bounds__(256)
void wavelet_softmax_kernel(const float* __restrict__ QK,
                            const float* __restrict__ Vm,
                            float* __restrict__ out) {
  constexpr int D = 1024, T = 256;
  __shared__ float sm[18];
  const size_t row = blockIdx.x;
  const float2* q2 = (const float2*)(QK + row * D);
  const float2* v2 = (const float2*)(Vm + row * D);
  float2* o2 = (float2*)(out + row * D);
  const int t = threadIdx.x;
  const float c = INV_SQRT2_F;

  float L[2], H[2], VL[2], VH[2];
#pragma unroll
  for (int i = 0; i < 2; ++i) {
    float2 q = q2[t + i * T];
    float2 v = v2[t + i * T];
    L[i]  = (q.x + q.y) * c;
    H[i]  = (q.y - q.x) * c;
    VL[i] = (v.x + v.y) * c;
    VH[i] = (v.y - v.x) * c;
  }

  // stage 1 softmax over the 512-wide L and H halves (dwt(q)+dwt(k)=dwt(q+k))
  float mL = fmaxf(L[0], L[1]), mH = fmaxf(H[0], H[1]);
  block_reduce2(mL, mH, true, sm);
  __syncthreads();
  float eL[2], eH[2], sL = 0.f, sH = 0.f;
#pragma unroll
  for (int i = 0; i < 2; ++i) {
    eL[i] = __expf(L[i] - mL); sL += eL[i];
    eH[i] = __expf(H[i] - mH); sH += eH[i];
  }
  block_reduce2(sL, sH, false, sm);
  __syncthreads();
  const float rL = 1.f / sL, rH = 1.f / sH;

  // stage 2 softmax over (softmax + v-half)
  float A2[2], B2[2];
#pragma unroll
  for (int i = 0; i < 2; ++i) {
    A2[i] = eL[i] * rL + VL[i];
    B2[i] = eH[i] * rH + VH[i];
  }
  float m2L = fmaxf(A2[0], A2[1]), m2H = fmaxf(B2[0], B2[1]);
  block_reduce2(m2L, m2H, true, sm);
  __syncthreads();
  float e2L[2], e2H[2], s2L = 0.f, s2H = 0.f;
#pragma unroll
  for (int i = 0; i < 2; ++i) {
    e2L[i] = __expf(A2[i] - m2L); s2L += e2L[i];
    e2H[i] = __expf(B2[i] - m2H); s2H += e2H[i];
  }
  block_reduce2(s2L, s2H, false, sm);
  const float r2L = 1.f / s2L, r2H = 1.f / s2H;

  // inverse Haar DWT + interleaved store
#pragma unroll
  for (int i = 0; i < 2; ++i) {
    float lo = e2L[i] * r2L;
    float ho = e2H[i] * r2H;
    float2 o;
    o.x = (lo - ho) * c;
    o.y = (lo + ho) * c;
    o2[t + i * T] = o;
  }
}

// ---------------------------------------------------------------------------
// Host-side launcher
// ---------------------------------------------------------------------------
extern "C" void kernel_launch(void* const* d_in, const int* in_sizes, int n_in,
                              void* d_out, int out_size, void* d_ws,
                              size_t ws_size, hipStream_t stream) {
  const float* x  = (const float*)d_in[0];
  const float* Wq = (const float*)d_in[1];
  const float* bq = (const float*)d_in[2];
  const float* Wk = (const float*)d_in[3];
  const float* bk = (const float*)d_in[4];
  const float* Wv = (const float*)d_in[5];
  const float* bv = (const float*)d_in[6];

  const int B = 8, S = 4096, D = 1024;
  const size_t M = (size_t)B * S;   // 32768 rows
  const size_t K = D, N = D;

  // workspace layout
  char* ws = (char*)d_ws;
  __bf16* xb  = (__bf16*)ws;  ws += M * K * sizeof(__bf16);         // 64 MB
  __bf16* wqk = (__bf16*)ws;  ws += (size_t)D * D * sizeof(__bf16); // 2 MB
  __bf16* wv  = (__bf16*)ws;  ws += (size_t)D * D * sizeof(__bf16); // 2 MB
  float*  bqk = (float*)ws;   ws += (size_t)D * sizeof(float);
  float*  QK  = (float*)ws;   ws += M * N * sizeof(float);          // 128 MB
  float*  Vm  = (float*)ws;                                         // 128 MB

  // 1) x -> bf16
  {
    size_t n4 = (M * K) / 4;
    dim3 grid((unsigned)((n4 + 255) / 256)), block(256);
    cvt_bf16_kernel<<<grid, block, 0, stream>>>(x, xb, n4);
  }
  // 2) weights: wqk = bf16(Wq+Wk), wv = bf16(Wv), bqk = bq+bk
  {
    size_t n4 = ((size_t)D * D) / 4;
    dim3 grid((unsigned)((n4 + 255) / 256)), block(256);
    prep_w_kernel<<<grid, block, 0, stream>>>(Wq, Wk, Wv, bq, bk,
                                              wqk, wv, bqk, n4, D / 4);
  }
  // 3) two WMMA GEMMs: QK = xb*wqk^T + bqk ; Vm = xb*wv^T + bv
  {
    dim3 grid((unsigned)(N / 128), (unsigned)(M / 128)), block(256);
    gemm_bf16_wmma_kernel<<<grid, block, 0, stream>>>(xb, wqk, bqk, QK,
                                                      (int)M, (int)N, (int)K);
    gemm_bf16_wmma_kernel<<<grid, block, 0, stream>>>(xb, wv, bv, Vm,
                                                      (int)M, (int)N, (int)K);
  }
  // 4) fused wavelet/softmax epilogue, one block per row
  {
    dim3 grid((unsigned)M), block(256);
    wavelet_softmax_kernel<<<grid, block, 0, stream>>>(QK, Vm, (float*)d_out);
  }
}
